// PCQSqueezeExcitation_71004399337901
// MI455X (gfx1250) — matrix operand order repaired and checked by reading, not compiled, not run on hardware
//
#include <hip/hip_runtime.h>
#include <hip/hip_bf16.h>

typedef __attribute__((ext_vector_type(16))) _Float16 v16h;
typedef __attribute__((ext_vector_type(8)))  float    v8f;
typedef __attribute__((ext_vector_type(4)))  float    f4;

constexpr int Bsz = 64;
constexpr int Cch = 256;
constexpr int HWc = 56 * 56;     // 3136
constexpr int HW4 = HWc / 4;     // 784
constexpr int Ssz = 64;
constexpr int Kcl = 8;

// ---- ordered-float <-> monotone unsigned key (for LDS atomicMin/Max) ----
__device__ __forceinline__ unsigned ford(float f) {
  unsigned u = __float_as_uint(f);
  return (u & 0x80000000u) ? ~u : (u | 0x80000000u);
}
__device__ __forceinline__ float funord(unsigned u) {
  unsigned v = (u & 0x80000000u) ? (u & 0x7FFFFFFFu) : ~u;
  return __uint_as_float(v);
}

// =====================================================================
// Kernel 1: one block per (b,c). Fused mean + min + max over H*W.
// Reads x exactly once (205 MB).
// =====================================================================
__global__ __launch_bounds__(256)
void pcq_pool_minmax(const float* __restrict__ x,
                     float* __restrict__ pooled,
                     float* __restrict__ xmn,
                     float* __restrict__ xmx) {
  __shared__ float ssum[256], smin[256], smax[256];
  const int bc = blockIdx.x;
  const int tid = threadIdx.x;
  const f4* xp = reinterpret_cast<const f4*>(x + (size_t)bc * HWc);
  float sum = 0.0f, mn = INFINITY, mx = -INFINITY;
  for (int i = tid; i < HW4; i += 256) {
    f4 v = xp[i];
    sum += (v.x + v.y) + (v.z + v.w);
    mn = fminf(mn, fminf(fminf(v.x, v.y), fminf(v.z, v.w)));
    mx = fmaxf(mx, fmaxf(fmaxf(v.x, v.y), fmaxf(v.z, v.w)));
  }
  ssum[tid] = sum; smin[tid] = mn; smax[tid] = mx;
  __syncthreads();
  for (int s = 128; s > 0; s >>= 1) {
    if (tid < s) {
      ssum[tid] += ssum[tid + s];
      smin[tid] = fminf(smin[tid], smin[tid + s]);
      smax[tid] = fmaxf(smax[tid], smax[tid + s]);
    }
    __syncthreads();
  }
  if (tid == 0) {
    pooled[bc] = ssum[0] * (1.0f / (float)HWc);
    xmn[bc] = smin[0];
    xmx[bc] = smax[0];
  }
}

// =====================================================================
// Kernel 2: single workgroup (8 waves). SE MLP with WMMA f16->f32,
// hardsigmoid scale, cluster min/max via LDS atomics, EMA, (s,z).
// =====================================================================
__global__ __launch_bounds__(256)
void pcq_se_stats(const float* __restrict__ pooled,
                  const float* __restrict__ xmn,
                  const float* __restrict__ xmx,
                  const float* __restrict__ w1, const float* __restrict__ b1,
                  const float* __restrict__ w2, const float* __restrict__ b2,
                  const float* __restrict__ act_range,
                  const int*   __restrict__ cluster_info,
                  float* __restrict__ scale,   // [B*C]
                  float* __restrict__ sb,      // [B]
                  float* __restrict__ zb) {    // [B]
  __shared__ _Float16 Hs[Bsz][Ssz];            // intermediate h (f16), 8 KB
  __shared__ unsigned rminU[Kcl], rmaxU[Kcl];
  __shared__ int      clusterOf[Bsz];
  __shared__ float    sArr[Kcl], zArr[Kcl];

  const int tid  = threadIdx.x;
  const int lane = tid & 31;
  const int wave = tid >> 5;

  if (tid < Kcl) { rminU[tid] = 0xFFFFFFFFu; rmaxU[tid] = 0u; }
  if (tid < Bsz) {
    int acc = 0, cl = 0;
    for (int k = 0; k < Kcl; ++k) {          // sample -> cluster from cluster_info
      int cnt = cluster_info[2 * k + 1];
      if (tid >= acc && tid < acc + cnt) cl = cluster_info[2 * k];
      acc += cnt;
    }
    clusterOf[tid] = cl;
  }
  __syncthreads();

  // wave32 WMMA 16x16x32 f16 fragment coordinates (per ISA layout tables)
  const int am  = lane & 15;                 // A: row M
  const int akb = (lane >> 4) << 3;          // A: K sub-base 0 / 8
  const int bn  = lane & 15;                 // B: col N
  const int bkb = (lane >> 4) << 4;          // unused: B's k reuses A's sub-base 0 / 8 (kb = ka)
  const int cm0 = (lane >> 4) << 3;          // C/D: m = v + cm0
  const int cn  = lane & 15;                 // C/D: n

  // ---------- GEMM1: h[64,64] = relu(pooled[64,256] @ w1[S,C]^T + b1) ----------
  for (int t = wave * 2; t < wave * 2 + 2; ++t) {   // 16 tiles, 2 per wave
    const int tm = t >> 2, tn = t & 3;
    v8f acc = {};
    for (int kk = 0; kk < 8; ++kk) {                // K=256 -> 8 steps of 32
      const int K0 = kk * 32;
      v16h a, b;
#pragma unroll
      for (int v = 0; v < 8; ++v) {
        const int ka = (v < 4) ? (akb + 2 * v) : (16 + akb + 2 * (v - 4));
        const float* ap = pooled + (tm * 16 + am) * Cch + K0 + ka;
        a[2 * v]     = (_Float16)ap[0];
        a[2 * v + 1] = (_Float16)ap[1];
        const int kb = ka;                          // B[k][n] = w1[s=n][c=k]
        const float* bp = w1 + (tn * 16 + bn) * Cch + K0 + kb;
        b[2 * v]     = (_Float16)bp[0];
        b[2 * v + 1] = (_Float16)bp[1];
      }
      acc = __builtin_amdgcn_wmma_f32_16x16x32_f16(false, a, false, b,
                                                   (short)0, acc, false, false);
    }
#pragma unroll
    for (int v = 0; v < 8; ++v) {
      const int bg = tm * 16 + cm0 + v;
      const int sg = tn * 16 + cn;
      float hv = acc[v] + b1[sg];
      Hs[bg][sg] = (_Float16)fmaxf(hv, 0.0f);
    }
  }
  __syncthreads();

  // ---------- GEMM2: g[64,256] = h[64,64] @ w2[C,S]^T + b2 -> hardsigmoid ----------
  for (int j = 0; j < 8; ++j) {                     // 64 tiles, 8 per wave
    const int t = wave * 8 + j;
    const int tm = t >> 4, tn = t & 15;
    v8f acc = {};
    for (int kk = 0; kk < 2; ++kk) {                // K=64 -> 2 steps of 32
      const int K0 = kk * 32;
      v16h a, b;
#pragma unroll
      for (int v = 0; v < 8; ++v) {
        const int ka = (v < 4) ? (akb + 2 * v) : (16 + akb + 2 * (v - 4));
        a[2 * v]     = Hs[tm * 16 + am][K0 + ka];
        a[2 * v + 1] = Hs[tm * 16 + am][K0 + ka + 1];
        const int kb = ka;                          // B[k][n] = w2[c=n][s=k]
        const float* bp = w2 + (tn * 16 + bn) * Ssz + K0 + kb;
        b[2 * v]     = (_Float16)bp[0];
        b[2 * v + 1] = (_Float16)bp[1];
      }
      acc = __builtin_amdgcn_wmma_f32_16x16x32_f16(false, a, false, b,
                                                   (short)0, acc, false, false);
    }
#pragma unroll
    for (int v = 0; v < 8; ++v) {
      const int bg = tm * 16 + cm0 + v;
      const int cg = tn * 16 + cn;
      float g  = acc[v] + b2[cg];
      float sc = fminf(fmaxf(g * (1.0f / 6.0f) + 0.5f, 0.0f), 1.0f);
      scale[bg * Cch + cg] = sc;
      // scale >= 0  =>  min/max over (h,w) of scale*x = scale * (min/max x)
      const float pmin = sc * xmn[bg * Cch + cg];
      const float pmax = sc * xmx[bg * Cch + cg];
      const int cl = clusterOf[bg];
      atomicMin(&rminU[cl], ford(pmin));
      atomicMax(&rmaxU[cl], ford(pmax));
    }
  }
  __syncthreads();

  if (tid < Kcl) {
    const float segmin = funord(rminU[tid]);
    const float segmax = funord(rmaxU[tid]);
    const float nmin = act_range[2 * tid]     * 0.995f + segmin * 0.005f;
    const float nmax = act_range[2 * tid + 1] * 0.995f + segmax * 0.005f;
    const float s = fmaxf((nmax - nmin) * (1.0f / 15.0f), 1e-8f);
    sArr[tid] = s;
    zArr[tid] = -rintf(nmin / s);
  }
  __syncthreads();
  if (tid < Bsz) {
    const int cl = clusterOf[tid];
    sb[tid] = sArr[cl];
    zb[tid] = zArr[cl];
  }
}

// =====================================================================
// Kernel 3: out = fakequant(scale[b,c] * x)  — 205 MB read + 205 MB write,
// non-temporal on both (no reuse).
// =====================================================================
__device__ __forceinline__ float fq1(float xv, float sc, float s, float inv, float z) {
  const float o = sc * xv;
  const float q = fminf(fmaxf(rintf(fmaf(o, inv, z)), 0.0f), 15.0f);
  return (q - z) * s;
}

__global__ __launch_bounds__(256)
void pcq_fakequant(const float* __restrict__ x,
                   const float* __restrict__ scale,
                   const float* __restrict__ sb,
                   const float* __restrict__ zb,
                   float* __restrict__ out) {
  const int i4 = blockIdx.x * 256 + threadIdx.x;
  constexpr int TOT4 = Bsz * Cch * HW4;
  if (i4 >= TOT4) return;
  const int bc = i4 / HW4;
  const int b  = bc / Cch;
  const float sc  = scale[bc];
  const float s   = sb[b];
  const float inv = 1.0f / s;
  const float z   = zb[b];
  f4 v = __builtin_nontemporal_load(reinterpret_cast<const f4*>(x) + i4);
  f4 o;
  o.x = fq1(v.x, sc, s, inv, z);
  o.y = fq1(v.y, sc, s, inv, z);
  o.z = fq1(v.z, sc, s, inv, z);
  o.w = fq1(v.w, sc, s, inv, z);
  __builtin_nontemporal_store(o, reinterpret_cast<f4*>(out) + i4);
}

extern "C" void kernel_launch(void* const* d_in, const int* in_sizes, int n_in,
                              void* d_out, int out_size, void* d_ws, size_t ws_size,
                              hipStream_t stream) {
  const float* x   = (const float*)d_in[0];
  const float* w1  = (const float*)d_in[1];
  const float* b1  = (const float*)d_in[2];
  const float* w2  = (const float*)d_in[3];
  const float* b2  = (const float*)d_in[4];
  const float* act = (const float*)d_in[5];
  const int*   ci  = (const int*)d_in[6];

  float* ws     = (float*)d_ws;
  float* pooled = ws;                    // [B*C]
  float* xmn    = ws + 1 * Bsz * Cch;    // [B*C]
  float* xmx    = ws + 2 * Bsz * Cch;    // [B*C]
  float* scale  = ws + 3 * Bsz * Cch;    // [B*C]
  float* sb     = ws + 4 * Bsz * Cch;    // [B]
  float* zb     = sb + Bsz;              // [B]
  float* out    = (float*)d_out;

  pcq_pool_minmax<<<Bsz * Cch, 256, 0, stream>>>(x, pooled, xmn, xmx);
  pcq_se_stats<<<1, 256, 0, stream>>>(pooled, xmn, xmx, w1, b1, w2, b2,
                                      act, ci, scale, sb, zb);
  constexpr int TOT4 = Bsz * Cch * HW4;
  pcq_fakequant<<<(TOT4 + 255) / 256, 256, 0, stream>>>(x, scale, sb, zb, out);
}
